// _CVGAECore_77979426226445
// MI455X (gfx1250) — matrix operand order repaired and checked by reading, not compiled, not run on hardware
//
#include <hip/hip_runtime.h>
#include <hip/hip_bf16.h>

// Problem constants (match reference).
#define NV   2048   // N nodes
#define NEDG 4096   // E edges
#define DD   256    // D
#define HEF  64     // HE
#define EPSF 1e-5f

typedef __attribute__((ext_vector_type(16))) _Float16 v16h;
typedef __attribute__((ext_vector_type(8)))  float    v8f;
typedef __attribute__((ext_vector_type(4)))  unsigned u32x4;
typedef __attribute__((ext_vector_type(8)))  unsigned u32x8;

union FragH { v16h v; unsigned u[8]; _Float16 h[16]; };
union FragF { v8f  v; float f[8]; };
union Pack4 { _Float16 h[4]; uint2 u2; };

__device__ __forceinline__ float apply_act(float x, int act) {
  switch (act) {
    case 1:  return x > 0.f ? x : 0.f;          // relu
    case 2:  return x > 0.f ? x : 0.01f * x;    // leaky_relu (jax default slope)
    case 3:  return tanhf(x);                   // tanh
    default: return x;
  }
}

#define TM 128
#define TN 128
#define TK 32

// ---------------------------------------------------------------------------
// Universal WMMA GEMM (f32 operands): C = act(A @ B + bias)
// ---------------------------------------------------------------------------
template <bool ALIGNED>
__global__ __launch_bounds__(256)
void gemm_wmma(const float* __restrict__ A, const float* __restrict__ B,
               const float* __restrict__ bias, float* __restrict__ C,
               int M, int N, int K, int act)
{
  __shared__ _Float16 sA[TM][TK + 8];   // [m][k], row stride 80B
  __shared__ _Float16 sB[TN][TK + 8];   // transposed: [n][k]

  const int tid  = threadIdx.x;
  const int lane = tid & 31;
  const int wid  = tid >> 5;
  const int wRow = wid >> 1;
  const int wCol = wid & 1;
  const int half = lane >> 4;
  const int l15  = lane & 15;

  const int bm = blockIdx.y * TM;
  const int bn = blockIdx.x * TN;

  FragF acc[2][4];
#pragma unroll
  for (int mi = 0; mi < 2; ++mi)
#pragma unroll
    for (int ni = 0; ni < 4; ++ni)
#pragma unroll
      for (int r = 0; r < 8; ++r) acc[mi][ni].f[r] = 0.f;

  const int kTiles = (K + TK - 1) / TK;
  const int ktPerZ = (kTiles + (int)gridDim.z - 1) / (int)gridDim.z;
  const int ktLo   = (int)blockIdx.z * ktPerZ;
  int ktHi = ktLo + ktPerZ; if (ktHi > kTiles) ktHi = kTiles;

  for (int kt = ktLo; kt < ktHi; ++kt) {
    const int k0 = kt * TK;

    if (ALIGNED) {
      float4 va[4], vb[4];
#pragma unroll
      for (int i = 0; i < 4; ++i) {
        int e = tid + i * 256;
        int r = e >> 3, c4 = (e & 7) << 2;
        va[i] = *reinterpret_cast<const float4*>(&A[(size_t)(bm + r) * K + k0 + c4]);
      }
#pragma unroll
      for (int i = 0; i < 4; ++i) {
        int e = tid + i * 256;
        int k = e >> 5, n4 = (e & 31) << 2;
        vb[i] = *reinterpret_cast<const float4*>(&B[(size_t)(k0 + k) * N + bn + n4]);
      }
#pragma unroll
      for (int i = 0; i < 4; ++i) {
        int e = tid + i * 256;
        int r = e >> 3, c4 = (e & 7) << 2;
        Pack4 t;
        t.h[0] = (_Float16)va[i].x; t.h[1] = (_Float16)va[i].y;
        t.h[2] = (_Float16)va[i].z; t.h[3] = (_Float16)va[i].w;
        *reinterpret_cast<uint2*>(&sA[r][c4]) = t.u2;
      }
#pragma unroll
      for (int i = 0; i < 4; ++i) {
        int e = tid + i * 256;
        int k = e >> 5, n4 = (e & 31) << 2;
        sB[n4 + 0][k] = (_Float16)vb[i].x;
        sB[n4 + 1][k] = (_Float16)vb[i].y;
        sB[n4 + 2][k] = (_Float16)vb[i].z;
        sB[n4 + 3][k] = (_Float16)vb[i].w;
      }
    } else {
#pragma unroll
      for (int i = 0; i < (TM * TK) / 256; ++i) {
        int e = tid + i * 256;
        int r = e >> 5, c = e & 31;
        int gm = bm + r, gk = k0 + c;
        float v = (gm < M && gk < K) ? A[(size_t)gm * K + gk] : 0.f;
        sA[r][c] = (_Float16)v;
      }
#pragma unroll
      for (int i = 0; i < (TK * TN) / 256; ++i) {
        int e = tid + i * 256;
        int n = e & 127, k = e >> 7;
        int gk = k0 + k, gn = bn + n;
        float v = (gk < K && gn < N) ? B[(size_t)gk * N + gn] : 0.f;
        sB[n][k] = (_Float16)v;
      }
    }

    if (kt + 1 < ktHi) {
      int pk = k0 + TK; if (pk >= K) pk = K - 1;
      int pm = bm + (tid & 127); if (pm >= M) pm = M - 1;
      int pn = bn + (tid & 127); if (pn >= N) pn = N - 1;
      __builtin_prefetch(&A[(size_t)pm * K + pk], 0, 1);
      __builtin_prefetch(&B[(size_t)pk * N + pn], 0, 1);
    }
    __syncthreads();

    FragH fa[2], fb[4];
#pragma unroll
    for (int mi = 0; mi < 2; ++mi) {
      const int m = wRow * 32 + mi * 16 + l15;
#pragma unroll
      for (int e = 0; e < 8; ++e) {
        const int kk = (e < 4) ? (2 * e + 8 * half) : (16 + 2 * (e - 4) + 8 * half);
        fa[mi].u[e] = *reinterpret_cast<const unsigned*>(&sA[m][kk]);
      }
    }
#pragma unroll
    for (int ni = 0; ni < 4; ++ni) {
      const int n = wCol * 64 + ni * 16 + l15;
#pragma unroll
      for (int e = 0; e < 8; ++e) {
        const int kk = (e < 4) ? (2 * e + 8 * half) : (16 + 2 * (e - 4) + 8 * half);
        fb[ni].u[e] = *reinterpret_cast<const unsigned*>(&sB[n][kk]);
      }
    }

#pragma unroll
    for (int mi = 0; mi < 2; ++mi)
#pragma unroll
      for (int ni = 0; ni < 4; ++ni)
        acc[mi][ni].v = __builtin_amdgcn_wmma_f32_16x16x32_f16(
            false, fa[mi].v, false, fb[ni].v, (short)0, acc[mi][ni].v,
            false, false);
    __syncthreads();
  }

  if (gridDim.z > 1) {
#pragma unroll
    for (int mi = 0; mi < 2; ++mi)
#pragma unroll
      for (int ni = 0; ni < 4; ++ni) {
        const int gn = bn + wCol * 64 + ni * 16 + l15;
        if (gn >= N) continue;
#pragma unroll
        for (int r = 0; r < 8; ++r) {
          const int gm = bm + wRow * 32 + mi * 16 + r + 8 * half;
          if (gm < M) atomicAdd(&C[(size_t)gm * N + gn], acc[mi][ni].f[r]);
        }
      }
  } else {
#pragma unroll
    for (int mi = 0; mi < 2; ++mi) {
#pragma unroll
      for (int ni = 0; ni < 4; ++ni) {
        const int gn = bn + wCol * 64 + ni * 16 + l15;
        if (gn >= N) continue;
        const float bv = bias ? bias[gn] : 0.f;
#pragma unroll
        for (int r = 0; r < 8; ++r) {
          const int gm = bm + wRow * 32 + mi * 16 + r + 8 * half;
          if (gm < M) C[(size_t)gm * N + gn] = apply_act(acc[mi][ni].f[r] + bv, act);
        }
      }
    }
  }
}

// ---------------------------------------------------------------------------
// TDM-fed WMMA GEMM for pre-converted f16 operands:
//   C[M,N] (+)= A16[M,K] @ B16t[N,K]^T      (B stored transposed, row-major)
// Requires M%128==0, N%128==0, K%(32*gridDim.z)==0. gridDim.z>1 -> atomicAdd
// into pre-zeroed C. Both LDS tiles are fetched by tensor_load_to_lds with a
// D# that pads 4 dwords after every 16 dwords -> 80B row stride in LDS,
// identical to the padded layout the fragment loads expect.
// ---------------------------------------------------------------------------
__global__ __launch_bounds__(256)
void gemm_h16_tdm(const _Float16* __restrict__ A, const _Float16* __restrict__ Bt,
                  float* __restrict__ C, int M, int N, int K)
{
  __shared__ _Float16 sA[TM][TK + 8];
  __shared__ _Float16 sB[TN][TK + 8];

  const int tid  = threadIdx.x;
  const int lane = tid & 31;
  const int wid  = tid >> 5;
  const int wRow = wid >> 1;
  const int wCol = wid & 1;
  const int half = lane >> 4;
  const int l15  = lane & 15;

  const int bm = blockIdx.y * TM;
  const int bn = blockIdx.x * TN;

  // Low 32 bits of the flat LDS-aperture address == LDS byte offset (ISA 10.2).
  const unsigned ldsA = (unsigned)(size_t)&sA[0][0];
  const unsigned ldsB = (unsigned)(size_t)&sB[0][0];

  // Loop-invariant D# group1:
  //  data_size=3 (8B units), pad_enable, pad_interval=3 (16 dwords),
  //  pad_amount=3 (4 dwords), tensor_dim0/1 = huge (tiles always in-bounds),
  //  tile_dim0=8 (64B row), tile_dim1=128 rows, dim0_stride = K/4 (8B units).
  u32x8 g1;
  g1[0] = (3u << 16) | (1u << 20) | (3u << 22) | (3u << 25);
  g1[1] = 0xFFFF0000u;                  // tensor_dim0[15:0] << 16
  g1[2] = 0xFFFFFFFFu;                  // tensor_dim0[31:16] | tensor_dim1[15:0]<<16
  g1[3] = (8u << 16) | 0xFFFFu;         // tile_dim0=8 | tensor_dim1[31:16]
  g1[4] = 128u;                         // tile_dim1=128, tile_dim2=0
  g1[5] = (unsigned)(K >> 2);           // tensor_dim0_stride low32 (8B units)
  g1[6] = 0u;
  g1[7] = 0u;

  FragF acc[2][4];
#pragma unroll
  for (int mi = 0; mi < 2; ++mi)
#pragma unroll
    for (int ni = 0; ni < 4; ++ni)
#pragma unroll
      for (int r = 0; r < 8; ++r) acc[mi][ni].f[r] = 0.f;

  const int kTiles = K / TK;
  const int ktPerZ = kTiles / (int)gridDim.z;
  const int ktLo   = (int)blockIdx.z * ktPerZ;
  const int ktHi   = ktLo + ktPerZ;

  for (int kt = ktLo; kt < ktHi; ++kt) {
    const int k0 = kt * TK;

    if (wid == 0) {
      unsigned long long ga = (unsigned long long)(size_t)(A  + (size_t)bm * K + k0);
      unsigned long long gb = (unsigned long long)(size_t)(Bt + (size_t)bn * K + k0);
      u32x4 g0a, g0b;
      g0a[0] = 1u;                              // count=1 (valid descriptor)
      g0a[1] = ldsA;                            // lds_addr
      g0a[2] = (unsigned)ga;                    // global_addr[31:0]
      g0a[3] = (unsigned)((ga >> 32) & 0x01FFFFFFull) | (2u << 30);  // addr hi | type=2
      g0b[0] = 1u;
      g0b[1] = ldsB;
      g0b[2] = (unsigned)gb;
      g0b[3] = (unsigned)((gb >> 32) & 0x01FFFFFFull) | (2u << 30);
      asm volatile("tensor_load_to_lds %0, %1" :: "s"(g0a), "s"(g1) : "memory");
      asm volatile("tensor_load_to_lds %0, %1" :: "s"(g0b), "s"(g1) : "memory");
      __builtin_amdgcn_s_wait_tensorcnt(0);
    }
    __syncthreads();

    FragH fa[2], fb[4];
#pragma unroll
    for (int mi = 0; mi < 2; ++mi) {
      const int m = wRow * 32 + mi * 16 + l15;
#pragma unroll
      for (int e = 0; e < 8; ++e) {
        const int kk = (e < 4) ? (2 * e + 8 * half) : (16 + 2 * (e - 4) + 8 * half);
        fa[mi].u[e] = *reinterpret_cast<const unsigned*>(&sA[m][kk]);
      }
    }
#pragma unroll
    for (int ni = 0; ni < 4; ++ni) {
      const int n = wCol * 64 + ni * 16 + l15;
#pragma unroll
      for (int e = 0; e < 8; ++e) {
        const int kk = (e < 4) ? (2 * e + 8 * half) : (16 + 2 * (e - 4) + 8 * half);
        fb[ni].u[e] = *reinterpret_cast<const unsigned*>(&sB[n][kk]);
      }
    }

#pragma unroll
    for (int mi = 0; mi < 2; ++mi)
#pragma unroll
      for (int ni = 0; ni < 4; ++ni)
        acc[mi][ni].v = __builtin_amdgcn_wmma_f32_16x16x32_f16(
            false, fa[mi].v, false, fb[ni].v, (short)0, acc[mi][ni].v,
            false, false);
    __syncthreads();
  }

  if (gridDim.z > 1) {
#pragma unroll
    for (int mi = 0; mi < 2; ++mi)
#pragma unroll
      for (int ni = 0; ni < 4; ++ni) {
        const int gn = bn + wCol * 64 + ni * 16 + l15;
#pragma unroll
        for (int r = 0; r < 8; ++r) {
          const int gm = bm + wRow * 32 + mi * 16 + r + 8 * half;
          atomicAdd(&C[(size_t)gm * N + gn], acc[mi][ni].f[r]);
        }
      }
  } else {
#pragma unroll
    for (int mi = 0; mi < 2; ++mi)
#pragma unroll
      for (int ni = 0; ni < 4; ++ni) {
        const int gn = bn + wCol * 64 + ni * 16 + l15;
#pragma unroll
        for (int r = 0; r < 8; ++r) {
          const int gm = bm + wRow * 32 + mi * 16 + r + 8 * half;
          C[(size_t)gm * N + gn] = acc[mi][ni].f[r];
        }
      }
  }
}

// ---------------------------------------------------------------------------
// Support kernels
// ---------------------------------------------------------------------------
__global__ void extract_idx(const float* __restrict__ vew, int* __restrict__ idx)
{
  int e = blockIdx.x * blockDim.x + threadIdx.x;
  if (e >= NEDG) return;
  int found = 0;
  for (int i = 0; i < NV; ++i) {
    if (vew[(size_t)i * NEDG + e] > 0.5f) { found = i; break; }
  }
  idx[e] = found;
}

__global__ void gather_uev(const float* __restrict__ hv, const float* __restrict__ he,
                           const int* __restrict__ u, const int* __restrict__ v,
                           float* __restrict__ uev)
{
  const int r  = blockIdx.x;
  const int er = (r < NEDG) ? r : r - NEDG;
  const int uu = (r < NEDG) ? u[er] : v[er];
  const int vv = (r < NEDG) ? v[er] : u[er];
  float* dst = uev + (size_t)r * (2 * DD + HEF);
  for (int j = threadIdx.x; j < DD;  j += blockDim.x) dst[j]            = hv[(size_t)uu * DD  + j];
  for (int j = threadIdx.x; j < HEF; j += blockDim.x) dst[DD + j]       = he[(size_t)er * HEF + j];
  for (int j = threadIdx.x; j < DD;  j += blockDim.x) dst[DD + HEF + j] = hv[(size_t)vv * DD  + j];
}

__global__ void attn_edge(const float* __restrict__ h, const float* __restrict__ w2,
                          const float* __restrict__ b2, float* __restrict__ ew)
{
  __shared__ float red[256];
  const int r = blockIdx.x;
  float p = 0.f;
  for (int j = threadIdx.x; j < DD; j += 256) p += h[(size_t)r * DD + j] * w2[j];
  red[threadIdx.x] = p;
  __syncthreads();
  for (int s = 128; s > 0; s >>= 1) {
    if (threadIdx.x < s) red[threadIdx.x] += red[threadIdx.x + s];
    __syncthreads();
  }
  if (threadIdx.x == 0) {
    float x = red[0] + b2[0];
    ew[r] = 1.f / (1.f + expf(-x));
  }
}

__global__ void scatter_adj(const int* __restrict__ u, const int* __restrict__ v,
                            const float* __restrict__ ew, float* __restrict__ adj)
{
  int r = blockIdx.x * blockDim.x + threadIdx.x;
  if (r >= 2 * NEDG) return;
  int er = (r < NEDG) ? r : r - NEDG;
  int uu = (r < NEDG) ? u[er] : v[er];
  int vv = (r < NEDG) ? v[er] : u[er];
  atomicAdd(&adj[(size_t)uu * NV + vv], ew[r]);
}

__global__ void row_sum(const float* __restrict__ adj, float* __restrict__ rs)
{
  __shared__ float red[256];
  const int i = blockIdx.x;
  float p = 0.f;
  for (int j = threadIdx.x; j < NV; j += 256) p += adj[(size_t)i * NV + j];
  red[threadIdx.x] = p;
  __syncthreads();
  for (int s = 128; s > 0; s >>= 1) {
    if (threadIdx.x < s) red[threadIdx.x] += red[threadIdx.x + s];
    __syncthreads();
  }
  if (threadIdx.x == 0) rs[i] = red[0];
}

__global__ void col_sum(const float* __restrict__ adj, float* __restrict__ cs)
{
  int j = blockIdx.x * blockDim.x + threadIdx.x;
  if (j >= NV) return;
  float p = 0.f;
  for (int i = 0; i < NV; ++i) p += adj[(size_t)i * NV + j];
  cs[j] = p;
}

// normalize and emit f16 adjacency for the TDM GEMM
__global__ void norm_adj16(const float* __restrict__ adj, const float* __restrict__ rs,
                           const float* __restrict__ cs, _Float16* __restrict__ adj16)
{
  size_t gid = (size_t)blockIdx.x * blockDim.x + threadIdx.x;
  if (gid >= (size_t)NV * NV) return;
  int i = (int)(gid / NV), j = (int)(gid % NV);
  float r = rs[i] < EPSF ? EPSF : rs[i];
  float c = cs[j] < EPSF ? EPSF : cs[j];
  adj16[gid] = (_Float16)(adj[gid] * rsqrtf(r) * rsqrtf(c));
}

// Xt16[c][r] = (f16) X[r][c]; R, C multiples of 32. 256 threads = 32x8 tile.
__global__ void transpose_f16(const float* __restrict__ X, _Float16* __restrict__ Xt,
                              int R, int Cc)
{
  __shared__ float t[32][33];
  const int rb = blockIdx.y * 32, cb = blockIdx.x * 32;
  const int tx = threadIdx.x & 31, ty = threadIdx.x >> 5;
  for (int i = ty; i < 32; i += 8) t[i][tx] = X[(size_t)(rb + i) * Cc + cb + tx];
  __syncthreads();
  for (int i = ty; i < 32; i += 8)
    Xt[(size_t)(cb + i) * R + rb + tx] = (_Float16)t[tx][i];
}

__global__ void add_vec(const float* __restrict__ a, const float* __restrict__ b,
                        float* __restrict__ o, int n)
{
  int i = blockIdx.x * blockDim.x + threadIdx.x;
  if (i < n) o[i] = a[i] + b[i];
}

__global__ void gru_update(const float* __restrict__ gi, const float* __restrict__ gh,
                           float* __restrict__ hidden)
{
  int gid = blockIdx.x * blockDim.x + threadIdx.x;
  if (gid >= NV * DD) return;
  int i = gid / DD, j = gid % DD;
  size_t b = (size_t)i * 3 * DD + j;
  float ir = gi[b], iz = gi[b + DD], in_ = gi[b + 2 * DD];
  float hr = gh[b], hz = gh[b + DD], hn  = gh[b + 2 * DD];
  float r = 1.f / (1.f + expf(-(ir + hr)));
  float z = 1.f / (1.f + expf(-(iz + hz)));
  float n = tanhf(in_ + r * hn);
  hidden[gid] = (1.f - z) * n + z * hidden[gid];
}

__global__ void concat2(const float* __restrict__ A, int ca,
                        const float* __restrict__ B, int cb,
                        float* __restrict__ O, int rows)
{
  int gid = blockIdx.x * blockDim.x + threadIdx.x;
  int cols = ca + cb;
  if (gid >= rows * cols) return;
  int i = gid / cols, j = gid % cols;
  O[gid] = (j < ca) ? A[(size_t)i * ca + j] : B[(size_t)i * cb + (j - ca)];
}

__global__ void make_sample(const float* __restrict__ post_out,
                            const float* __restrict__ eps, float* __restrict__ s)
{
  int gid = blockIdx.x * blockDim.x + threadIdx.x;
  if (gid >= NV * DD) return;
  int i = gid / DD, j = gid % DD;
  float mu = post_out[(size_t)i * 2 * DD + j];
  float ls = post_out[(size_t)i * 2 * DD + DD + j];
  s[gid] = mu + expf(0.5f * ls) * eps[gid];
}

__global__ void kld_kernel(const float* __restrict__ prior, const float* __restrict__ post,
                           float* __restrict__ kz, float* __restrict__ k0)
{
  __shared__ float rz[256], r0[256];
  const int i = blockIdx.x;
  const int j = threadIdx.x;
  float pm = prior[(size_t)i * 2 * DD + j];
  float pl = prior[(size_t)i * 2 * DD + DD + j];
  float qm = post [(size_t)i * 2 * DD + j];
  float ql = post [(size_t)i * 2 * DD + DD + j];
  float var0 = expf(ql), var1 = expf(pl);
  float a = (var0 + EPSF) / (var1 + EPSF);
  float b = (pm - qm) * (pm - qm) / (var1 + EPSF);
  float c = logf((var1 + EPSF) / (var0 + EPSF) + EPSF);
  rz[j] = a + b - 1.f + c;
  r0[j] = var1 + pm * pm - 1.f - pl;
  __syncthreads();
  for (int s = 128; s > 0; s >>= 1) {
    if (j < s) { rz[j] += rz[j + s]; r0[j] += r0[j + s]; }
    __syncthreads();
  }
  if (j == 0) { kz[i] = 0.5f * rz[0]; k0[i] = 0.5f * r0[0]; }
}

// ---------------------------------------------------------------------------
// Host-side orchestration
// ---------------------------------------------------------------------------
struct MpnnParams {
  const float *att_W1, *att_b1, *att_W2, *att_b2;
  const float *gcn_W, *gcn_b;
  const float *gru_Wih, *gru_Whh, *gru_bih, *gru_bhh;
  const float *rm_W1, *rm_b1, *rm_W2, *rm_b2;
};

static MpnnParams get_params(void* const* d_in, int b) {
  MpnnParams p;
  p.att_W1  = (const float*)d_in[b + 0];
  p.att_b1  = (const float*)d_in[b + 1];
  p.att_W2  = (const float*)d_in[b + 2];
  p.att_b2  = (const float*)d_in[b + 3];
  p.gcn_W   = (const float*)d_in[b + 4];
  p.gcn_b   = (const float*)d_in[b + 5];
  p.gru_Wih = (const float*)d_in[b + 6];
  p.gru_Whh = (const float*)d_in[b + 7];
  p.gru_bih = (const float*)d_in[b + 8];
  p.gru_bhh = (const float*)d_in[b + 9];
  p.rm_W1   = (const float*)d_in[b + 10];
  p.rm_b1   = (const float*)d_in[b + 11];
  p.rm_W2   = (const float*)d_in[b + 12];
  p.rm_b2   = (const float*)d_in[b + 13];
  return p;
}

struct WS {
  int *u, *v;
  float *uev, *h_att, *ew, *adj, *rs, *cs;
  _Float16 *adj16, *ht16;
  float *hidden, *msgpre, *msg, *gi, *gh;
  float *hvp, *Xc, *X2, *h2;
  float *prior_out, *post_out, *sample;
};

static void launch_gemm(hipStream_t s, const float* A, const float* B,
                        const float* bias, float* C, int M, int N, int K, int act)
{
  dim3 g((N + TN - 1) / TN, (M + TM - 1) / TM, 1);
  if ((M % TM) == 0 && (N % TN) == 0 && (K % TK) == 0)
    gemm_wmma<true ><<<g, 256, 0, s>>>(A, B, bias, C, M, N, K, act);
  else
    gemm_wmma<false><<<g, 256, 0, s>>>(A, B, bias, C, M, N, K, act);
}

static void run_mpnn(hipStream_t s, const MpnnParams& P, const float* hv,
                     const float* he, const float* sample, const WS& w, float* out)
{
  // edge attention
  gather_uev<<<2 * NEDG, 256, 0, s>>>(hv, he, w.u, w.v, w.uev);
  launch_gemm(s, w.uev, P.att_W1, P.att_b1, w.h_att, 2 * NEDG, DD, 2 * DD + HEF, /*leaky*/2);
  attn_edge<<<2 * NEDG, 256, 0, s>>>(w.h_att, P.att_W2, P.att_b2, w.ew);

  // adjacency build + symmetric normalization (emit f16 for the TDM GEMM)
  hipMemsetAsync(w.adj, 0, (size_t)NV * NV * sizeof(float), s);
  scatter_adj<<<(2 * NEDG + 255) / 256, 256, 0, s>>>(w.u, w.v, w.ew, w.adj);
  row_sum<<<NV, 256, 0, s>>>(w.adj, w.rs);
  col_sum<<<(NV + 255) / 256, 256, 0, s>>>(w.adj, w.cs);
  norm_adj16<<<(int)(((size_t)NV * NV + 255) / 256), 256, 0, s>>>(w.adj, w.rs, w.cs, w.adj16);

  // hidden init: hvs = hv (+ sample)
  if (sample)
    add_vec<<<(NV * DD + 255) / 256, 256, 0, s>>>(hv, sample, w.hidden, NV * DD);
  else
    hipMemcpyAsync(w.hidden, hv, (size_t)NV * DD * sizeof(float),
                   hipMemcpyDeviceToDevice, s);

  // 4 GCN + GRU layers
  for (int l = 0; l < 4; ++l) {
    // msgpre = adj16 @ hidden: TDM-fed f16 GEMM, split-K x8 (32 -> 256 WGs)
    transpose_f16<<<dim3(DD / 32, NV / 32), 256, 0, s>>>(w.hidden, w.ht16, NV, DD);
    hipMemsetAsync(w.msgpre, 0, (size_t)NV * DD * sizeof(float), s);
    gemm_h16_tdm<<<dim3(DD / TN, NV / TM, 8), 256, 0, s>>>(w.adj16, w.ht16,
                                                           w.msgpre, NV, DD, NV);
    launch_gemm(s, w.msgpre, P.gcn_W + (size_t)l * DD * DD, P.gcn_b + l * DD,
                w.msg, NV, DD, DD, /*relu*/1);
    launch_gemm(s, w.msg,    P.gru_Wih, P.gru_bih, w.gi, NV, 3 * DD, DD, 0);
    launch_gemm(s, w.hidden, P.gru_Whh, P.gru_bhh, w.gh, NV, 3 * DD, DD, 0);
    gru_update<<<(NV * DD + 255) / 256, 256, 0, s>>>(w.gi, w.gh, w.hidden);
  }

  // remap MLP
  concat2<<<((NV * 2 * DD) + 255) / 256, 256, 0, s>>>(hv, DD, w.hidden, DD, w.X2, NV);
  launch_gemm(s, w.X2, P.rm_W1, P.rm_b1, w.h2, NV, DD, 2 * DD, /*leaky*/2);
  launch_gemm(s, w.h2, P.rm_W2, P.rm_b2, out, NV, 2 * DD, DD, 0);
}

extern "C" void kernel_launch(void* const* d_in, const int* in_sizes, int n_in,
                              void* d_out, int out_size, void* d_ws, size_t ws_size,
                              hipStream_t stream)
{
  (void)in_sizes; (void)n_in; (void)out_size; (void)ws_size;

  const float* hv   = (const float*)d_in[0];
  const float* he   = (const float*)d_in[1];
  const float* pos  = (const float*)d_in[2];
  const float* eps  = (const float*)d_in[3];
  const float* vew1 = (const float*)d_in[4];
  const float* vew2 = (const float*)d_in[5];
  const float* vp_w = (const float*)d_in[6];
  const float* vp_b = (const float*)d_in[7];
  MpnnParams prior = get_params(d_in, 8);
  MpnnParams post  = get_params(d_in, 22);
  MpnnParams pred  = get_params(d_in, 36);

  char* base = (char*)d_ws;
  size_t off = 0;
  auto alloc = [&](size_t bytes) -> void* {
    void* p = base + off;
    off += (bytes + 255) & ~(size_t)255;
    return p;
  };
  WS w;
  w.u         = (int*)      alloc((size_t)NEDG * sizeof(int));
  w.v         = (int*)      alloc((size_t)NEDG * sizeof(int));
  w.uev       = (float*)    alloc((size_t)2 * NEDG * (2 * DD + HEF) * 4);
  w.h_att     = (float*)    alloc((size_t)2 * NEDG * DD * 4);
  w.ew        = (float*)    alloc((size_t)2 * NEDG * 4);
  w.adj       = (float*)    alloc((size_t)NV * NV * 4);
  w.adj16     = (_Float16*) alloc((size_t)NV * NV * 2);
  w.ht16      = (_Float16*) alloc((size_t)DD * NV * 2);
  w.rs        = (float*)    alloc((size_t)NV * 4);
  w.cs        = (float*)    alloc((size_t)NV * 4);
  w.hidden    = (float*)    alloc((size_t)NV * DD * 4);
  w.msgpre    = (float*)    alloc((size_t)NV * DD * 4);
  w.msg       = (float*)    alloc((size_t)NV * DD * 4);
  w.gi        = (float*)    alloc((size_t)NV * 3 * DD * 4);
  w.gh        = (float*)    alloc((size_t)NV * 3 * DD * 4);
  w.hvp       = (float*)    alloc((size_t)NV * DD * 4);
  w.Xc        = (float*)    alloc((size_t)NV * (DD + 3) * 4);
  w.X2        = (float*)    alloc((size_t)NV * 2 * DD * 4);
  w.h2        = (float*)    alloc((size_t)NV * DD * 4);
  w.prior_out = (float*)    alloc((size_t)NV * 2 * DD * 4);
  w.post_out  = (float*)    alloc((size_t)NV * 2 * DD * 4);
  w.sample    = (float*)    alloc((size_t)NV * DD * 4);

  float* out_x  = (float*)d_out;                       // [N, 2D]
  float* out_kz = out_x + (size_t)NV * 2 * DD;         // [N]
  float* out_k0 = out_kz + NV;                         // [N]

  extract_idx<<<(NEDG + 255) / 256, 256, 0, stream>>>(vew1, w.u);
  extract_idx<<<(NEDG + 255) / 256, 256, 0, stream>>>(vew2, w.v);

  run_mpnn(stream, prior, hv, he, nullptr, w, w.prior_out);

  concat2<<<((NV * (DD + 3)) + 255) / 256, 256, 0, stream>>>(hv, DD, pos, 3, w.Xc, NV);
  launch_gemm(stream, w.Xc, vp_w, vp_b, w.hvp, NV, DD, DD + 3, /*tanh*/3);

  run_mpnn(stream, post, w.hvp, he, nullptr, w, w.post_out);
  make_sample<<<(NV * DD + 255) / 256, 256, 0, stream>>>(w.post_out, eps, w.sample);

  run_mpnn(stream, pred, hv, he, w.sample, w, out_x);

  kld_kernel<<<NV, 256, 0, stream>>>(w.prior_out, w.post_out, out_kz, out_k0);
}
